// CriticNetwork_89713276879307
// MI455X (gfx1250) — matrix operand order repaired and checked by reading, not compiled, not run on hardware
//
#include <hip/hip_runtime.h>

#define WAVES 2
#define NAG 16     // agents per batch
#define AD  32     // action dim
#define IND 128    // obs input dim
#define DD  128    // attention dim
#define HD  32     // MLP hidden dim

typedef __attribute__((ext_vector_type(16))) __bf16 v16bf;
typedef __attribute__((ext_vector_type(8)))  __bf16 v8bf;
typedef __attribute__((ext_vector_type(8)))  float  v8f;
typedef __attribute__((ext_vector_type(4)))  float  v4f;
typedef __attribute__((ext_vector_type(2)))  unsigned v2u;
typedef __attribute__((ext_vector_type(4)))  unsigned v4u;

// strides chosen so every row base is 16-byte aligned (for b128 LDS ops)
#define OBS_S 136   // 272 B/row
#define KQ_S  136
#define H_S   40    // 80 B/row
#define AL_S  24    // 48 B/row

struct alignas(16) WaveScratch {
  __bf16 obs[NAG][OBS_S];      // 4352 B
  __bf16 kq[2][NAG][KQ_S];     // 8704 B  (k, q row-major)
  __bf16 vcm[DD][NAG];         // 4096 B  (v column-major: vcm[d][j])
  __bf16 h[NAG][H_S];          // 1280 B
  __bf16 alpha[NAG][AL_S];     // 768 B
  float  s1[NAG];
  float  pdot[NAG];
  float  adot[NAG];
  float  nz[NAG][NAG];
  float  nzsum[NAG];
};

// hardware tanh (gfx1250 transcendental) with branch-free fallback
__device__ __forceinline__ float fast_tanh(float x) {
#if __has_builtin(__builtin_amdgcn_tanhf)
  return __builtin_amdgcn_tanhf(x);
#elif __has_builtin(__builtin_amdgcn_tanh_f32)
  return __builtin_amdgcn_tanh_f32(x);
#else
  float xc = fminf(fmaxf(x, -15.0f), 15.0f);
  float t = __expf(2.0f * xc);
  return (t - 1.0f) / (t + 1.0f);
#endif
}

// A-fragment K-pair index: VGPR p(0-7): K=hi*16+2p
__device__ __forceinline__ int a_kpair(int p, int hi) {
  return hi * 16 + (p << 1);
}

// Build 16-bit A fragment with eight 32-bit LDS reads (pairs are consecutive K)
__device__ __forceinline__ v16bf load_a_pairs(const __bf16* row, int k0, int hi) {
  union { v16bf v; unsigned u[8]; } t;
#pragma unroll
  for (int p = 0; p < 8; ++p)
    t.u[p] = *(const unsigned*)(row + k0 + a_kpair(p, hi));
  return t.v;
}

// Load a pre-packed B fragment: two coalesced 16-byte chunks, 256 elems apart
__device__ __forceinline__ v16bf load_packed(const __bf16* p) {
  v8bf lo = *(const v8bf*)(p);
  v8bf hi8 = *(const v8bf*)(p + 256);
  return __builtin_shufflevector(lo, hi8, 0, 1, 2, 3, 4, 5, 6, 7,
                                 8, 9, 10, 11, 12, 13, 14, 15);
}

// Load 16 contiguous bf16 from LDS (16-byte aligned) as a B fragment
__device__ __forceinline__ v16bf load_b_row(const __bf16* p) {
  v8bf lo = *(const v8bf*)(p);
  v8bf hi8 = *(const v8bf*)(p + 8);
  return __builtin_shufflevector(lo, hi8, 0, 1, 2, 3, 4, 5, 6, 7,
                                 8, 9, 10, 11, 12, 13, 14, 15);
}

// ---- one-shot weight pre-pack: f32 -> bf16 in per-lane B-fragment order ----
__global__ void pack_weights_kernel(
    const float* __restrict__ Wk1, const float* __restrict__ Wq1,
    const float* __restrict__ Wv1, const float* __restrict__ Wk2,
    const float* __restrict__ Wq2, const float* __restrict__ Wv2,
    __bf16* __restrict__ out) {
  int p = blockIdx.x * blockDim.x + threadIdx.x;
  if (p >= 6 * 4096) return;
  const float* W1s[3] = {Wk1, Wq1, Wv1};
  const float* W2s[3] = {Wk2, Wq2, Wv2};
  int region = p / (3 * 4096);
  int r = p % (3 * 4096);
  int h = r / 4096;
  int q = r % 4096;
  int f = q / 512;                 // fragment id
  int o = q % 512;
  int c = o / 256;                 // chunk 0/1
  int lane = (o % 256) / 8;
  int e = c * 8 + (o % 8);         // element within fragment
  int hi = lane >> 4, m = lane & 15;
  float val;
  if (region == 0) {
    int kt = f >> 1, nt = f & 1;   // W1: [128 x 32]
    val = W1s[h][(kt * 32 + hi * 16 + e) * HD + nt * 16 + m];
  } else {                         // W2: [32 x 128], frag = output tile t
    val = W2s[h][(hi * 16 + e) * DD + f * 16 + m];
  }
  out[p] = (__bf16)val;
}

__global__ __launch_bounds__(WAVES * 32) void critic_fused_kernel(
    const float* __restrict__ obs, const float* __restrict__ policies,
    const float* __restrict__ actions, const float* __restrict__ weights,
    const float* __restrict__ noise,
    const float* __restrict__ bk1, const float* __restrict__ bk2,
    const float* __restrict__ bq1, const float* __restrict__ bq2,
    const float* __restrict__ bv1, const float* __restrict__ bv2,
    const float* __restrict__ Wval, const float* __restrict__ bval,
    const __bf16* __restrict__ wpack,
    float* __restrict__ out_x, float* __restrict__ out_w) {
  __shared__ WaveScratch ws_all[WAVES];
  const int wid  = threadIdx.x >> 5;
  const int lane = threadIdx.x & 31;
  const int m    = lane & 15;
  const int hi   = lane >> 4;
  WaveScratch& ws = ws_all[wid];
  const int bat = blockIdx.x * WAVES + wid;

  // ---- stage obs tile [16 x 128] as bf16: b128 loads, 8B packed LDS stores ----
  {
    const v4f* src4 = (const v4f*)(obs + (size_t)bat * NAG * IND);
#pragma unroll
    for (int g = lane; g < NAG * IND / 4; g += 32) {
      int r = g >> 5, c4 = (g & 31) << 2;   // 32 float4 groups per row
      v4f f = src4[g];
      union { __bf16 b[4]; v2u u; } t;
      t.b[0] = (__bf16)f[0]; t.b[1] = (__bf16)f[1];
      t.b[2] = (__bf16)f[2]; t.b[3] = (__bf16)f[3];
      *(v2u*)&ws.obs[r][c4] = t.u;
    }
  }
  __syncthreads();

  const float* B1s[3] = {bk1, bq1, bv1};
  const float* B2s[3] = {bk2, bq2, bv2};

  // ---- k/q/v MLPs via WMMA bf16 with pre-packed weights ----
#pragma unroll
  for (int hsel = 0; hsel < 3; ++hsel) {
    const float* B1 = B1s[hsel];
    const float* B2 = B2s[hsel];
    const __bf16* w1p = wpack + hsel * 4096 + lane * 8;
    const __bf16* w2p = wpack + 3 * 4096 + hsel * 4096 + lane * 8;

    // layer1: [16x128] @ [128x32] -> two 16x16 tiles, K-loop 4x32
    v8f acc0 = {}, acc1 = {};
#pragma unroll
    for (int kt = 0; kt < IND / 32; ++kt) {
      v16bf a = load_a_pairs(&ws.obs[m][0], kt * 32, hi);
      v16bf bL = load_packed(w1p + (kt * 2 + 0) * 512);
      v16bf bH = load_packed(w1p + (kt * 2 + 1) * 512);
      acc0 = __builtin_amdgcn_wmma_f32_16x16x32_bf16(false, a, false, bL, (short)0, acc0, false, false);
      acc1 = __builtin_amdgcn_wmma_f32_16x16x32_bf16(false, a, false, bH, (short)0, acc1, false, false);
    }
#pragma unroll
    for (int r = 0; r < 8; ++r) {
      int row = r + 8 * hi;
      ws.h[row][m]      = (__bf16)fast_tanh(acc0[r] + B1[m]);
      ws.h[row][16 + m] = (__bf16)fast_tanh(acc1[r] + B1[16 + m]);
    }
    __syncthreads();

    // layer2: [16x32] @ [32x128] -> 8 tiles, single K=32 step each
    v16bf a2 = load_a_pairs(&ws.h[m][0], 0, hi);
#pragma unroll
    for (int t = 0; t < DD / 16; ++t) {
      v16bf bb = load_packed(w2p + t * 512);
      v8f c = {};
      c = __builtin_amdgcn_wmma_f32_16x16x32_bf16(false, a2, false, bb, (short)0, c, false, false);
      int col = t * 16 + m;
      if (hsel < 2) {
        // k, q: row-major scatter (rows needed per-lane later)
#pragma unroll
        for (int r = 0; r < 8; ++r)
          ws.kq[hsel][r + 8 * hi][col] = (__bf16)(c[r] + B2[col]);
      } else {
        // v: column-major, 8 consecutive rows per lane -> one b128 store
        union { __bf16 b[8]; v4u u; } pk;
#pragma unroll
        for (int r = 0; r < 8; ++r) pk.b[r] = (__bf16)(c[r] + B2[col]);
        *(v4u*)&ws.vcm[col][8 * hi] = pk.u;
      }
    }
    __syncthreads();
  }

  // ---- scores[i][j] = q[i].k[j]/sqrt(D); softmax over j ----
  v8f sc = {};
#pragma unroll
  for (int kt = 0; kt < DD / 32; ++kt) {
    v16bf a  = load_a_pairs(&ws.kq[1][m][0], kt * 32, hi);        // q rows
    v16bf bb = load_b_row(&ws.kq[0][m][kt * 32 + hi * 16]);       // B[d][j]=k[j][d]
    sc = __builtin_amdgcn_wmma_f32_16x16x32_bf16(false, a, false, bb, (short)0, sc, false, false);
  }
  const float scl = 0.08838834764831845f;  // 1/sqrt(128)
#pragma unroll
  for (int r = 0; r < 8; ++r) {
    float s = sc[r] * scl;                 // row i=r+8*hi across 16 lanes (j=m)
    float mx = s;
#pragma unroll
    for (int msk = 1; msk <= 8; msk <<= 1) mx = fmaxf(mx, __shfl_xor(mx, msk, 32));
    float ex = __expf(s - mx);
    float sum = ex;
#pragma unroll
    for (int msk = 1; msk <= 8; msk <<= 1) sum += __shfl_xor(sum, msk, 32);
    float al = ex / sum;
    int row = r + 8 * hi;
    ws.alpha[row][m] = (__bf16)al;
    out_w[(size_t)(bat * NAG + row) * NAG + m] = al;   // weights_preproc output
  }
  __syncthreads();

  // ---- obs_proc = alpha @ v (K padded to 32) fused with s1[j] reduction ----
  v16bf aA;
  {
    union { v16bf v; unsigned u[8]; } t;
#pragma unroll
    for (int p = 0; p < 8; ++p) {
      int k = a_kpair(p, hi);
      t.u[p] = (k < NAG) ? *(const unsigned*)&ws.alpha[m][k] : 0u;
    }
    aA = t.v;
  }
  float s1acc[8];
#pragma unroll
  for (int r = 0; r < 8; ++r) s1acc[r] = 0.0f;
#pragma unroll
  for (int t = 0; t < DD / 16; ++t) {
    // B[k=j][n=dcol] = v[j][dcol] = vcm[dcol][j]: contiguous for hi=0, zero pad hi=1
    v16bf bb = {};
    if (hi == 0) bb = load_b_row(&ws.vcm[t * 16 + m][0]);
    v8f c = {};
    c = __builtin_amdgcn_wmma_f32_16x16x32_bf16(false, aA, false, bb, (short)0, c, false, false);
    float wv = Wval[t * 16 + m];
#pragma unroll
    for (int r = 0; r < 8; ++r) s1acc[r] += c[r] * wv;
  }
#pragma unroll
  for (int r = 0; r < 8; ++r) {
    float v = s1acc[r];
#pragma unroll
    for (int msk = 1; msk <= 8; msk <<= 1) v += __shfl_xor(v, msk, 32);
    if (m == 0) ws.s1[r + 8 * hi] = v;
  }

  // ---- collapsed tail: dots with u = Wval[128:160] (b128 loads) ----
  const float* uvec = Wval + DD;
  v4f u4[AD / 4];
#pragma unroll
  for (int k4 = 0; k4 < AD / 4; ++k4) u4[k4] = ((const v4f*)uvec)[k4];
  {
    const v4f* src = (const v4f*)((hi ? actions : policies) + (size_t)(bat * NAG + m) * AD);
    float dot = 0.0f;
#pragma unroll
    for (int k4 = 0; k4 < AD / 4; ++k4) {
      v4f f = src[k4];
      dot += f[0] * u4[k4][0] + f[1] * u4[k4][1] + f[2] * u4[k4][2] + f[3] * u4[k4][3];
    }
    if (hi) ws.adot[m] = dot; else ws.pdot[m] = dot;
  }
#pragma unroll
  for (int t = 0; t < 8; ++t) {
    int idx = lane + 32 * t;
    int i = idx >> 4, j = idx & 15;
    const v4f* np = (const v4f*)(noise + ((size_t)(bat * NAG + i) * NAG + j) * AD);
    float dz = 0.0f;
#pragma unroll
    for (int k4 = 0; k4 < AD / 4; ++k4) {
      v4f f = np[k4];
      dz += f[0] * u4[k4][0] + f[1] * u4[k4][1] + f[2] * u4[k4][2] + f[3] * u4[k4][3];
    }
    ws.nz[i][j] = dz;
  }
  __syncthreads();
  if (lane < NAG) {
    float s = 0.0f;
#pragma unroll
    for (int j = 0; j < NAG; ++j) s += ws.nz[lane][j];
    ws.nzsum[lane] = s;
  }
  __syncthreads();

  const float w  = weights[0];
  const float bv = bval[0];
  float Psum = 0.0f, Asum = 0.0f;
#pragma unroll
  for (int j = 0; j < NAG; ++j) { Psum += ws.pdot[j]; Asum += ws.adot[j]; }
#pragma unroll
  for (int t = 0; t < 8; ++t) {
    int idx = lane + 32 * t;
    int i = idx >> 4, j = idx & 15;
    float zd = w * ws.adot[j] + (1.0f - w) * ws.pdot[j] + ws.nz[i][j];
    float zs = w * Asum + (1.0f - w) * Psum + ws.nzsum[i];
    float x  = ws.s1[j] + (ws.pdot[j] - zd + zs) * (1.0f / NAG) + bv;
    out_x[(size_t)(bat * NAG + i) * NAG + j] = x;
  }
}

extern "C" void kernel_launch(void* const* d_in, const int* in_sizes, int n_in,
                              void* d_out, int out_size, void* d_ws, size_t ws_size,
                              hipStream_t stream) {
  (void)n_in; (void)out_size; (void)ws_size;
  const float* obs      = (const float*)d_in[0];
  const float* policies = (const float*)d_in[1];
  const float* actions  = (const float*)d_in[2];
  const float* weights  = (const float*)d_in[3];
  const float* noise    = (const float*)d_in[4];
  const float* Wk1 = (const float*)d_in[5];  const float* bk1 = (const float*)d_in[6];
  const float* Wk2 = (const float*)d_in[7];  const float* bk2 = (const float*)d_in[8];
  const float* Wq1 = (const float*)d_in[9];  const float* bq1 = (const float*)d_in[10];
  const float* Wq2 = (const float*)d_in[11]; const float* bq2 = (const float*)d_in[12];
  const float* Wv1 = (const float*)d_in[13]; const float* bv1 = (const float*)d_in[14];
  const float* Wv2 = (const float*)d_in[15]; const float* bv2 = (const float*)d_in[16];
  const float* Wval = (const float*)d_in[17]; const float* bval = (const float*)d_in[18];

  const int nodes = in_sizes[0] / IND;     // 32768
  const int B     = nodes / NAG;           // 2048
  float* out_x = (float*)d_out;
  float* out_w = out_x + (size_t)nodes * NAG;
  __bf16* wpack = (__bf16*)d_ws;           // 24576 bf16 = 48 KB

  pack_weights_kernel<<<(6 * 4096 + 255) / 256, 256, 0, stream>>>(
      Wk1, Wq1, Wv1, Wk2, Wq2, Wv2, wpack);

  dim3 grid(B / WAVES), block(WAVES * 32);
  critic_fused_kernel<<<grid, block, 0, stream>>>(
      obs, policies, actions, weights, noise,
      bk1, bk2, bq1, bq2, bv1, bv2, Wval, bval, wpack, out_x, out_w);
}
